// OTStyleModel_64037962383619
// MI455X (gfx1250) — matrix operand order repaired and checked by reading, not compiled, not run on hardware
//
#include <hip/hip_runtime.h>
#include <hip/hip_bf16.h>
#include <math.h>

typedef __attribute__((ext_vector_type(16))) _Float16 v16h;
typedef __attribute__((ext_vector_type(8)))  float    v8f;

namespace {
constexpr int   B_    = 64;
constexpr int   L_    = 128;
constexpr int   HID_  = 768;
constexpr int   CDIM_ = 256;
constexpr int   SDIM_ = 64;
constexpr int   FEAT_ = CDIM_ + SDIM_ + 1;   // 321
constexpr int   NIT_  = 30;
constexpr float EPS_  = 0.05f;

// LDS layout
constexpr int SH_STRIDE = HID_ + 8;             // f16 elems/row (+16B pad -> conflict-free b128 frag loads)
constexpr int CM_STRIDE = L_ + 4;               // f32 elems/row (bank-skewed row scans)
constexpr int SH_BYTES  = L_ * SH_STRIDE * 2;   // 198656
constexpr int CM_BYTES  = L_ * CM_STRIDE * 4;   // 67584
constexpr int AUX_FLOATS = 2 * HID_ /*rep*/ + 324 /*feat(pad)*/ + 7 * L_ + 3 * 256 + 16;
constexpr size_t LDS_BYTES = (size_t)SH_BYTES + CM_BYTES + (size_t)AUX_FLOATS * 4; // ~280 KB < 320 KB/WGP

struct alignas(16) Q16 { unsigned int a, b, c, d; };
union FragU { Q16 q[2]; v16h v; };
} // namespace

__global__ __launch_bounds__(256, 1)
void ot_sinkhorn_kernel(const float* __restrict__ H,
                        const int*   __restrict__ tt,
                        const int*   __restrict__ am,
                        const float* __restrict__ W_c,
                        const float* __restrict__ b_c,
                        const float* __restrict__ W_s,
                        const float* __restrict__ b_s,
                        const float* __restrict__ gate,
                        const float* __restrict__ ln_w,
                        const float* __restrict__ ln_b,
                        const float* __restrict__ W_cls,
                        const float* __restrict__ b_cls,
                        float*       __restrict__ out)
{
    extern __shared__ char smem[];
    _Float16* sH  = (_Float16*)smem;
    float* Cm   = (float*)(smem + SH_BYTES);
    float* rep  = (float*)(smem + SH_BYTES + CM_BYTES);
    float* feat = rep  + 2 * HID_;
    float* m0fL = feat + 324;
    float* m1fL = m0fL + L_;
    float* la   = m1fL + L_;
    float* lb   = la   + L_;
    float* fpot = lb   + L_;
    float* gpot = fpot + L_;
    float* sqL  = gpot + L_;
    float* pm   = sqL  + L_;
    float* ps   = pm   + 256;
    float* red  = ps   + 256;
    float* scal = red  + 256;

    const int b     = blockIdx.x;
    const int tid   = threadIdx.x;
    const int lane  = tid & 31;
    const int wave  = tid >> 5;
    const int lhalf = lane >> 4;   // 0/1
    const int lmod  = lane & 15;

    const float* Hb = H + (size_t)b * L_ * HID_;

    // ---- Phase 0: H (f32) -> sH (f16 LDS); masks ----
    for (int r = wave; r < L_; r += 8) {
        const float* src = Hb + r * HID_;
        _Float16*    dst = sH + r * SH_STRIDE;
        for (int c = lane; c < HID_; c += 32)
            dst[c] = (_Float16)src[c];
    }
    if (tid < L_) {
        int a = am[b * L_ + tid];
        int t = tt[b * L_ + tid];
        m0fL[tid] = (a == 1 && t == 0) ? 1.f : 0.f;
        m1fL[tid] = (a == 1 && t == 1) ? 1.f : 0.f;
    }
    __syncthreads();
    if (tid == 0) {
        float c0 = 0.f, c1 = 0.f;
        for (int t = 0; t < L_; ++t) { c0 += m0fL[t]; c1 += m1fL[t]; }
        scal[0] = fmaxf(c0, 1.f);
        scal[1] = fmaxf(c1, 1.f);
    }
    __syncthreads();
    const float n0c = scal[0];
    const float n1c = scal[1];

    // ---- Phase 1: Gram = sH * sH^T via WMMA (wave w -> 16-row output strip) ----
    {
        v8f zero = {0.f, 0.f, 0.f, 0.f, 0.f, 0.f, 0.f, 0.f};
        v8f acc[8];
        #pragma unroll
        for (int c = 0; c < 8; ++c) acc[c] = zero;

        const int arow = wave * 16 + lmod;
        for (int k0 = 0; k0 < HID_; k0 += 32) {
            FragU af;
            // A 16x32 f16: lane l holds M=l%16; K = kb..kb+7 then 16+kb..16+kb+7, kb=(l/16)*8
            const _Float16* abase = sH + arow * SH_STRIDE + k0 + lhalf * 8;
            af.q[0] = *(const Q16*)(abase);
            af.q[1] = *(const Q16*)(abase + 16);
            #pragma unroll
            for (int c = 0; c < 8; ++c) {
                FragU bf;
                // B 32x16 f16: lane l holds N=l%16; K = kb..kb+7 then 16+kb..16+kb+7, kb=(l/16)*8
                const _Float16* bbase = sH + (c * 16 + lmod) * SH_STRIDE + k0 + lhalf * 8;
                bf.q[0] = *(const Q16*)(bbase);
                bf.q[1] = *(const Q16*)(bbase + 16);
                acc[c] = __builtin_amdgcn_wmma_f32_16x16x32_f16(
                    false, af.v, false, bf.v, (short)0, acc[c], false, false);
            }
        }
        // D layout: VGPR v -> M = v + 8*(l/16), N = l%16
        #pragma unroll
        for (int c = 0; c < 8; ++c) {
            float* drow = Cm + (wave * 16 + 8 * lhalf) * CM_STRIDE + c * 16 + lmod;
            #pragma unroll
            for (int v = 0; v < 8; ++v)
                drow[v * CM_STRIDE] = acc[c][v];
        }
    }
    __syncthreads();

    // ---- Phase 2: sq from Gram diagonal, log-weights ----
    if (tid < L_) {
        sqL[tid] = Cm[tid * CM_STRIDE + tid];
        la[tid]  = (m0fL[tid] > 0.f) ? -__logf(n0c) : -INFINITY;
        lb[tid]  = (m1fL[tid] > 0.f) ? -__logf(n1c) : -INFINITY;
    }
    __syncthreads();
    // cost transform: Cm = sqrt(clip(sq_i + sq_j - 2 G, 1e-6))
    for (int idx = tid; idx < L_ * L_; idx += 256) {
        int i = idx >> 7, j = idx & (L_ - 1);
        float gv = Cm[i * CM_STRIDE + j];
        float d2 = sqL[i] + sqL[j] - 2.f * gv;
        Cm[i * CM_STRIDE + j] = sqrtf(fmaxf(d2, 1e-6f));
    }

    // ---- Phase 3: rep = [cls, m0 - m1] (f32 from global for precision) ----
    for (int d = tid; d < HID_; d += 256)
        rep[d] = Hb[d];                                  // cls = token 0
    for (int d = tid; d < HID_; d += 256) {
        float s0 = 0.f, s1 = 0.f;
        for (int t = 0; t < L_; ++t) {
            float h = Hb[t * HID_ + d];
            s0 += m0fL[t] * h;
            s1 += m1fL[t] * h;
        }
        rep[HID_ + d] = s0 / n0c - s1 / n1c;
    }
    __syncthreads();  // rep ready, Cm transform complete

    // ---- Phase 4: fused features (C, S projections) ----
    const float gsig = 1.f / (1.f + __expf(-gate[0]));
    {
        const float* wrow = W_c + (size_t)tid * (2 * HID_);
        float acc = b_c[tid];
        for (int k = 0; k < 2 * HID_; ++k) acc += rep[k] * wrow[k];
        feat[tid] = acc * (1.f - gsig);
    }
    if (tid < SDIM_) {
        const float* wrow = W_s + (size_t)tid * (2 * HID_);
        float acc = b_s[tid];
        for (int k = 0; k < 2 * HID_; ++k) acc += rep[k] * wrow[k];
        feat[CDIM_ + tid] = acc * gsig;
    }

    // ---- Phase 5: Sinkhorn divergence, 3 runs, 30 iters each ----
    const float inv_eps = 1.f / EPS_;
    const int   row  = tid & (L_ - 1);
    const int   half = tid >> 7;
    const int   jlo  = half * 64, jhi = jlo + 64;
    const float* Crow = Cm + row * CM_STRIDE;   // Cm symmetric: column scan == row scan

    for (int run = 0; run < 3; ++run) {
        const float* wa = (run == 2) ? lb : la;
        const float* wb = (run == 1) ? la : lb;
        if (tid < L_) { fpot[tid] = 0.f; gpot[tid] = 0.f; }
        __syncthreads();

        for (int it = 0; it < NIT_; ++it) {
            // f_i = -eps * lse_j( wb[j] + (g[j] - C[i][j])/eps )
            {
                float m = -INFINITY, s = 0.f;
                for (int j = jlo; j < jhi; ++j) {
                    float w = wb[j];
                    if (w > -1e30f) {
                        float v = w + (gpot[j] - Crow[j]) * inv_eps;
                        if (v > m) { s = s * __expf(m - v) + 1.f; m = v; }
                        else       { s += __expf(v - m); }
                    }
                }
                pm[tid] = m; ps[tid] = s;
            }
            __syncthreads();
            if (tid < L_) {
                float m0 = pm[tid], s0 = ps[tid];
                float m1 = pm[tid + 128], s1 = ps[tid + 128];
                float lse;
                if      (m1 == -INFINITY) lse = m0 + __logf(s0);
                else if (m0 == -INFINITY) lse = m1 + __logf(s1);
                else {
                    float M = fmaxf(m0, m1);
                    lse = M + __logf(s0 * __expf(m0 - M) + s1 * __expf(m1 - M));
                }
                fpot[tid] = -EPS_ * lse;
            }
            __syncthreads();
            // g_j = -eps * lse_i( wa[i] + (f[i] - C[i][j])/eps )   (row == column j)
            {
                float m = -INFINITY, s = 0.f;
                for (int i = jlo; i < jhi; ++i) {
                    float w = wa[i];
                    if (w > -1e30f) {
                        float v = w + (fpot[i] - Crow[i]) * inv_eps;
                        if (v > m) { s = s * __expf(m - v) + 1.f; m = v; }
                        else       { s += __expf(v - m); }
                    }
                }
                pm[tid] = m; ps[tid] = s;
            }
            __syncthreads();
            if (tid < L_) {
                float m0 = pm[tid], s0 = ps[tid];
                float m1 = pm[tid + 128], s1 = ps[tid + 128];
                float lse;
                if      (m1 == -INFINITY) lse = m0 + __logf(s0);
                else if (m0 == -INFINITY) lse = m1 + __logf(s1);
                else {
                    float M = fmaxf(m0, m1);
                    lse = M + __logf(s0 * __expf(m0 - M) + s1 * __expf(m1 - M));
                }
                gpot[tid] = -EPS_ * lse;
            }
            __syncthreads();
        }
        // dual value <a,f> + <b,g>
        if (tid < L_)
            red[tid] = __expf(wa[tid]) * fpot[tid] + __expf(wb[tid]) * gpot[tid];
        __syncthreads();
        for (int off = 64; off > 0; off >>= 1) {
            if (tid < off) red[tid] += red[tid + off];
            __syncthreads();
        }
        if (tid == 0) scal[2 + run] = red[0];
        __syncthreads();
    }

    // ---- Phase 6: d_ot, LayerNorm, classifier ----
    if (tid == 0)
        feat[FEAT_ - 1] = scal[2] - 0.5f * (scal[3] + scal[4]);
    __syncthreads();

    const bool hiv = (tid + 256) < FEAT_;
    float x0 = feat[tid];
    float x1 = hiv ? feat[tid + 256] : 0.f;

    red[tid] = x0 + x1;
    __syncthreads();
    for (int off = 128; off > 0; off >>= 1) {
        if (tid < off) red[tid] += red[tid + off];
        __syncthreads();
    }
    const float mu = red[0] / (float)FEAT_;
    __syncthreads();

    float d0 = x0 - mu;
    float d1 = hiv ? (x1 - mu) : 0.f;
    red[tid] = d0 * d0 + d1 * d1;
    __syncthreads();
    for (int off = 128; off > 0; off >>= 1) {
        if (tid < off) red[tid] += red[tid + off];
        __syncthreads();
    }
    const float var  = red[0] / (float)FEAT_;
    const float rstd = rsqrtf(var + 1e-5f);
    __syncthreads();

    float h0 = d0 * rstd * ln_w[tid] + ln_b[tid];
    float h1 = hiv ? (d1 * rstd * ln_w[tid + 256] + ln_b[tid + 256]) : 0.f;

    red[tid] = h0 * W_cls[tid] + (hiv ? h1 * W_cls[tid + 256] : 0.f);
    __syncthreads();
    for (int off = 128; off > 0; off >>= 1) {
        if (tid < off) red[tid] += red[tid + off];
        __syncthreads();
    }
    if (tid == 0) scal[5] = red[0];
    __syncthreads();

    red[tid] = h0 * W_cls[FEAT_ + tid] + (hiv ? h1 * W_cls[FEAT_ + tid + 256] : 0.f);
    __syncthreads();
    for (int off = 128; off > 0; off >>= 1) {
        if (tid < off) red[tid] += red[tid + off];
        __syncthreads();
    }
    if (tid == 0) {
        out[b * 2 + 0] = scal[5] + b_cls[0];
        out[b * 2 + 1] = red[0]  + b_cls[1];
    }
}

extern "C" void kernel_launch(void* const* d_in, const int* in_sizes, int n_in,
                              void* d_out, int out_size, void* d_ws, size_t ws_size,
                              hipStream_t stream) {
    (void)in_sizes; (void)n_in; (void)out_size; (void)d_ws; (void)ws_size;
    const float* H     = (const float*)d_in[0];
    const int*   tt    = (const int*)  d_in[1];
    const int*   am    = (const int*)  d_in[2];
    const float* W_c   = (const float*)d_in[3];
    const float* b_c   = (const float*)d_in[4];
    const float* W_s   = (const float*)d_in[5];
    const float* b_s   = (const float*)d_in[6];
    const float* gate  = (const float*)d_in[7];
    const float* ln_w  = (const float*)d_in[8];
    const float* ln_b  = (const float*)d_in[9];
    const float* W_cls = (const float*)d_in[10];
    const float* b_cls = (const float*)d_in[11];
    float* out = (float*)d_out;

    // Opt in to large dynamic LDS (host-side attribute set; no alloc/sync — graph-capture safe).
    (void)hipFuncSetAttribute((const void*)ot_sinkhorn_kernel,
                              hipFuncAttributeMaxDynamicSharedMemorySize,
                              (int)LDS_BYTES);

    ot_sinkhorn_kernel<<<dim3(B_), dim3(256), LDS_BYTES, stream>>>(
        H, tt, am, W_c, b_c, W_s, b_s, gate, ln_w, ln_b, W_cls, b_cls, out);
}
